// Mix9Net_52553219834298
// MI455X (gfx1250) — matrix operand order repaired and checked by reading, not compiled, not run on hardware
//
#include <hip/hip_runtime.h>
#include <hip/hip_bf16.h>

typedef __attribute__((ext_vector_type(16))) _Float16 v16h;
typedef __attribute__((ext_vector_type(8)))  _Float16 v8h;
typedef __attribute__((ext_vector_type(8)))  float    v8f;

#define DM    128
#define NPIX  225
#define NPAD  240
#define PLANE (NPAD * DM)   // halfs per activation plane (61440 B)

__device__ __forceinline__ float silu_f(float v) {
  return v / (1.0f + __expf(-v));
}

// ---------------------------------------------------------------------------
// A (weight) fragment for v_wmma_f32_16x16x32_f16.
// W layout: [oc][ic] f16 row-major (128 ic per row).
// Per ISA: lane L holds row M = L&15; element e = 8g+w maps to
//          K = 32*kstep + 16*g + 8*(L>>4) + w
// ---------------------------------------------------------------------------
__device__ __forceinline__ v16h load_a_frag(const _Float16* __restrict__ W,
                                            int mbase, int lane, int kstep) {
  const int oc = mbase + (lane & 15);
  const int c0 = kstep * 32 + ((lane >> 4) << 3);
  const _Float16* row = W + oc * DM;
  v8h lo = *(const v8h*)(row + c0);
  v8h hi = *(const v8h*)(row + c0 + 16);
  v16h a;
#pragma unroll
  for (int i = 0; i < 8; ++i) { a[i] = lo[i]; a[i + 8] = hi[i]; }
  return a;
}

// ---------------------------------------------------------------------------
// One fused channel-mix layer over the LDS-resident plane.
//   TAPS  : 1 (conv1x1) or 3 (directional conv, shifts (t-1)*(dy,dx))
//   SILU  : apply silu to GEMM output
//   RESID : add residual plane Rs (same rows this wave owns)
//   GOUT  : write fp32 to global (final layer) instead of LDS plane D
// Plane layout: pixel-major f16, plane[p*128 + c].
// Pad pixels 225..239 are kept zero for the whole kernel, so out-of-range /
// out-of-image B columns are handled purely by address selection (no data
// masking): invalid source pixels read pad pixel 225 (all-zero channels).
// Epilogue stores are guarded with p < 225 so pad pixels stay zero.
// ---------------------------------------------------------------------------
template <int TAPS, bool SILU, bool RESID, bool GOUT>
__device__ __forceinline__ void gemm_layer(
    const _Float16* __restrict__ S, _Float16* __restrict__ D,
    const _Float16* __restrict__ Wg,   // f16 weights, [tap][oc][ic]
    const float* __restrict__ bias,
    const _Float16* __restrict__ Rs,
    float* __restrict__ gout,
    int dy, int dx, int mtile, int n0, int n1, int lane) {
  const int mbase = mtile * 16;
  const int half_ = lane >> 4;
  const int col   = lane & 15;

  v16h af[TAPS][4];
#pragma unroll
  for (int t = 0; t < TAPS; ++t)
#pragma unroll
    for (int s = 0; s < 4; ++s)
      af[t][s] = load_a_frag(Wg + t * DM * DM, mbase, lane, s);

  float bs[8];
#pragma unroll
  for (int r = 0; r < 8; ++r) bs[r] = bias[mbase + half_ * 8 + r];

  for (int n = n0; n < n1; ++n) {
    const int p = n * 16 + col;
    const int y = p / 15;
    const int x = p - y * 15;
    v8f c = {};
#pragma unroll
    for (int t = 0; t < TAPS; ++t) {
      int sp;
      if (TAPS == 3) {
        const int sh = t - 1;
        const int sy = y + sh * dy;
        const int sx = x + sh * dx;
        const bool valid =
            (p < NPIX) & ((unsigned)sy < 15u) & ((unsigned)sx < 15u);
        sp = valid ? (sy * 15 + sx) : NPIX;  // pixel 225 is all-zero
      } else {
        sp = p;  // pad pixels hold zeros; result discarded by guarded store
      }
      const _Float16* bp = S + sp * DM;
#pragma unroll
      for (int s = 0; s < 4; ++s) {
        v8h blo = *(const v8h*)(bp + s * 32 + half_ * 8);
        v8h bhi = *(const v8h*)(bp + s * 32 + half_ * 8 + 16);
        v16h bv;
#pragma unroll
        for (int i = 0; i < 8; ++i) { bv[i] = blo[i]; bv[i + 8] = bhi[i]; }
        c = __builtin_amdgcn_wmma_f32_16x16x32_f16(
            false, af[t][s], false, bv, (short)0, c, false, false);
      }
    }
    // epilogue (guarded: pad pixels must stay zero in the planes)
    if (p < NPIX) {
      float vals[8];
#pragma unroll
      for (int r = 0; r < 8; ++r) {
        float v = c[r] + bs[r];
        if (SILU) v = silu_f(v);
        vals[r] = v;
      }
      if (RESID) {
        v8h rv = *(const v8h*)(Rs + p * DM + mbase + half_ * 8);
#pragma unroll
        for (int r = 0; r < 8; ++r) vals[r] += (float)rv[r];
      }
      if (GOUT) {
#pragma unroll
        for (int r = 0; r < 8; ++r)
          gout[(mbase + half_ * 8 + r) * NPIX + p] = vals[r];
      } else {
        v8h o;
#pragma unroll
        for (int r = 0; r < 8; ++r) o[r] = (_Float16)vals[r];
        *(v8h*)(D + p * DM + mbase + half_ * 8) = o;
      }
    }
  }
}

// ---------------------------------------------------------------------------
// Fully fused network: one workgroup per (batch, direction) image.
// ---------------------------------------------------------------------------
__global__ __launch_bounds__(256) void mix9_fused(
    const float* __restrict__ x_in, const float* __restrict__ w_d0,
    const float* __restrict__ b_d0,
    const _Float16* __restrict__ wd16, const float* __restrict__ res_bd,
    const _Float16* __restrict__ w116, const float* __restrict__ res_b1,
    const _Float16* __restrict__ c0w1, const float* __restrict__ c0b1,
    const _Float16* __restrict__ c0w2, const float* __restrict__ c0b2,
    const _Float16* __restrict__ wf16, const float* __restrict__ bf,
    float* __restrict__ out) {
  extern __shared__ _Float16 smem[];
  _Float16* Xp = smem;          // network state plane
  _Float16* Tp = smem + PLANE;  // temp plane

  const int img = blockIdx.x;
  const int b = img >> 2;
  const int d = img & 3;
  // DIRS = (0,1),(1,0),(1,1),(1,-1)
  const int dy = (d == 0) ? 0 : 1;
  const int dx = (d == 1) ? 0 : ((d == 3) ? -1 : 1);

  const int tid  = threadIdx.x;
  const int lane = tid & 31;
  const int wv   = tid >> 5;

  // zero both planes (pad pixels 225..239 must stay zero for the whole run)
  for (int i = tid; i < (2 * PLANE) / 8; i += 256) {
    v8h z = {};
    *(v8h*)(smem + i * 8) = z;
  }
  __syncthreads();

  // ---- layer 0: X = silu(dconv0(x)), Cin = 2 (scalar VALU) ----
  const float* xb = x_in + b * 2 * NPIX;
  for (int idx = tid; idx < NPIX * DM; idx += 256) {
    const int oc = idx & 127;
    const int p = idx >> 7;
    const int y = p / 15, x = p - y * 15;
    float acc = b_d0[oc];
#pragma unroll
    for (int k = 0; k < 3; ++k) {
      const int sy = y + (k - 1) * dy;
      const int sx = x + (k - 1) * dx;
      if ((unsigned)sy < 15u && (unsigned)sx < 15u) {
        const int sp = sy * 15 + sx;
        acc += w_d0[oc * 6 + k] * xb[sp];            // ic = 0
        acc += w_d0[oc * 6 + 3 + k] * xb[NPIX + sp]; // ic = 1
      }
    }
    Xp[p * DM + oc] = (_Float16)silu_f(acc);
  }
  __syncthreads();

  // ---- 4 residual blocks ----
  for (int l = 0; l < 4; ++l) {
    // T = silu(dconv(X))
    gemm_layer<3, true, false, false>(Xp, Tp, wd16 + l * 3 * DM * DM,
                                      res_bd + l * DM, nullptr, nullptr,
                                      dy, dx, wv, 0, 15, lane);
    __syncthreads();
    // X = silu(conv1x1(T)) + X   (each wave owns its 16 output rows)
    gemm_layer<1, true, true, false>(Tp, Xp, w116 + l * DM * DM,
                                     res_b1 + l * DM, Xp, nullptr,
                                     dy, dx, wv, 0, 15, lane);
    __syncthreads();
  }

  // ---- head: X = silu(1x1(silu(1x1(X)))) + X ----
  gemm_layer<1, true, false, false>(Xp, Tp, c0w1, c0b1, nullptr, nullptr,
                                    dy, dx, wv, 0, 15, lane);
  __syncthreads();
  gemm_layer<1, true, true, false>(Tp, Xp, c0w2, c0b2, Xp, nullptr,
                                   dy, dx, wv, 0, 15, lane);
  __syncthreads();

  // ---- final: out = conv1x1(X, wf), M = 64, fp32 to global ----
  float* gout = out + (size_t)img * 64 * NPIX;
  gemm_layer<1, false, false, true>(Xp, nullptr, wf16, bf, nullptr, gout,
                                    dy, dx, wv & 3, (wv >> 2) * 8,
                                    (wv >> 2) ? 15 : 8, lane);
}

// ---------------------------------------------------------------------------
// Weight preparation: fp32 -> f16 (dconv weights re-laid out [l][k][oc][ic])
// ---------------------------------------------------------------------------
__global__ void cvt_wd(const float* __restrict__ in, _Float16* __restrict__ out) {
  const int i = blockIdx.x * 256 + threadIdx.x;
  if (i >= 4 * 3 * DM * DM) return;
  const int ic = i & 127;
  int j = i >> 7;
  const int oc = j & 127;
  j >>= 7;            // j = l*3 + k
  const int k = j % 3;
  const int l = j / 3;
  out[i] = (_Float16)in[((l * DM + oc) * DM + ic) * 3 + k];
}

__global__ void cvt_f32_f16(const float* __restrict__ in,
                            _Float16* __restrict__ out, int n) {
  const int i = blockIdx.x * 256 + threadIdx.x;
  if (i < n) out[i] = (_Float16)in[i];
}

// ---------------------------------------------------------------------------
extern "C" void kernel_launch(void* const* d_in, const int* in_sizes, int n_in,
                              void* d_out, int out_size, void* d_ws, size_t ws_size,
                              hipStream_t stream) {
  const float* x      = (const float*)d_in[0];
  const float* w_d0   = (const float*)d_in[1];
  const float* b_d0   = (const float*)d_in[2];
  const float* res_wd = (const float*)d_in[3];
  const float* res_bd = (const float*)d_in[4];
  const float* res_w1 = (const float*)d_in[5];
  const float* res_b1 = (const float*)d_in[6];
  const float* c0_w1  = (const float*)d_in[7];
  const float* c0_b1  = (const float*)d_in[8];
  const float* c0_w2  = (const float*)d_in[9];
  const float* c0_b2  = (const float*)d_in[10];
  const float* wf     = (const float*)d_in[11];
  const float* bf     = (const float*)d_in[12];
  float* out = (float*)d_out;

  // workspace: f16 weights
  _Float16* ws    = (_Float16*)d_ws;
  _Float16* wd16  = ws;                       // 4*3*128*128 = 196608
  _Float16* w116  = wd16 + 4 * 3 * DM * DM;   // 4*128*128   =  65536
  _Float16* c0w1h = w116 + 4 * DM * DM;       //             =  16384
  _Float16* c0w2h = c0w1h + DM * DM;          //             =  16384
  _Float16* wf16  = c0w2h + DM * DM;          // 64*128      =   8192

  cvt_wd<<<(4 * 3 * DM * DM + 255) / 256, 256, 0, stream>>>(res_wd, wd16);
  cvt_f32_f16<<<(4 * DM * DM + 255) / 256, 256, 0, stream>>>(res_w1, w116, 4 * DM * DM);
  cvt_f32_f16<<<(DM * DM + 255) / 256, 256, 0, stream>>>(c0_w1, c0w1h, DM * DM);
  cvt_f32_f16<<<(DM * DM + 255) / 256, 256, 0, stream>>>(c0_w2, c0w2h, DM * DM);
  cvt_f32_f16<<<(64 * DM + 255) / 256, 256, 0, stream>>>(wf, wf16, 64 * DM);

  const size_t lds_bytes = (size_t)2 * PLANE * sizeof(_Float16); // 122880
  (void)hipFuncSetAttribute((const void*)mix9_fused,
                            hipFuncAttributeMaxDynamicSharedMemorySize,
                            (int)lds_bytes);

  mix9_fused<<<1024 * 4, 256, lds_bytes, stream>>>(
      x, w_d0, b_d0, wd16, res_bd, w116, res_b1, c0w1h, c0_b1, c0w2h, c0_b2,
      wf16, bf, out);
}
